// MaskedSelfAttention_4088808866189
// MI455X (gfx1250) — matrix operand order repaired and checked by reading, not compiled, not run on hardware
//
#include <hip/hip_runtime.h>

// Problem constants
#define BB   2
#define SS   2048
#define DD   1024
#define HH   16
#define HDIM 64

typedef __attribute__((ext_vector_type(16))) __bf16 v16bf;
typedef __attribute__((ext_vector_type(8)))  float  v8f;

// ---------- bf16 bit helpers (RNE) ----------
__device__ __forceinline__ unsigned short f2bf(float f) {
  unsigned u = __float_as_uint(f);
  u += 0x7FFFu + ((u >> 16) & 1u);
  return (unsigned short)(u >> 16);
}
__device__ __forceinline__ float bf2f(unsigned short h) {
  return __uint_as_float(((unsigned)h) << 16);
}
__device__ __forceinline__ v8f zero_v8f() {
  v8f z;
#pragma unroll
  for (int i = 0; i < 8; ++i) z[i] = 0.0f;
  return z;
}

// ---------- gfx1250 async global->LDS copy (ASYNCcnt tracked) ----------
__device__ __forceinline__ void async_b128(unsigned short* lds_dst,
                                           const unsigned short* gsrc) {
  // VDST = per-lane LDS byte address (low 32 bits of flat shared pointer),
  // VADDR = 64-bit global address, GV mode (SADDR = off/NULL).
  asm volatile("global_load_async_to_lds_b128 %0, %1, off"
               :: "v"((unsigned)(size_t)lds_dst), "v"(gsrc)
               : "memory");
}
__device__ __forceinline__ void wait_async0() {
  asm volatile("s_wait_asynccnt 0" ::: "memory");
}

// ---------- WMMA fragment loaders (per CDNA5 ISA 7.12.2 layouts) ----------
// A-frag (16x32 bf16, row-major [m][k] in LDS): lane m = lane%16,
// VGPR v<4: k = 2v + 8*(lane/16); v>=4: k = 16 + 2(v-4) + 8*(lane/16)
__device__ __forceinline__ v16bf load_a_frag(const unsigned short* p, int stride,
                                             int row, int kof) {
  const int lane = threadIdx.x & 31;
  const int hf = lane >> 4;
  union { v16bf v; unsigned u[8]; } f;
  const unsigned short* r0 = p + row * stride + kof + 8 * hf;
#pragma unroll
  for (int i = 0; i < 4; ++i) f.u[i] = *(const unsigned*)(r0 + 2 * i);
#pragma unroll
  for (int i = 4; i < 8; ++i) f.u[i] = *(const unsigned*)(r0 + 16 + 2 * (i - 4));
  return f.v;
}
// B-frag (32x16 bf16, LDS stored transposed [n][k]): lane n = lane%16,
// VGPR v<4: k = 2v + 8*(lane/16); v>=4: k = 16 + 2(v-4) + 8*(lane/16)
__device__ __forceinline__ v16bf load_b_frag(const unsigned short* p, int stride,
                                             int nrow, int kof) {
  const int lane = threadIdx.x & 31;
  const int hf = lane >> 4;
  union { v16bf v; unsigned u[8]; } f;
  const unsigned short* r0 = p + nrow * stride + kof + 8 * hf;
#pragma unroll
  for (int i = 0; i < 4; ++i) f.u[i] = *(const unsigned*)(r0 + 2 * i);
#pragma unroll
  for (int i = 4; i < 8; ++i) f.u[i] = *(const unsigned*)(r0 + 16 + 2 * (i - 4));
  return f.v;
}

__device__ __forceinline__ v8f wmma_bf16(v16bf a, v16bf b, v8f c) {
  return __builtin_amdgcn_wmma_f32_16x16x32_bf16(false, a, false, b, (short)0, c,
                                                 false, false);
}

// ---------- Kernel 0: f32 -> bf16 convert ----------
__global__ __launch_bounds__(256) void k_cvt(const float* __restrict__ src,
                                             unsigned short* __restrict__ dst) {
  int i = (blockIdx.x * 256 + threadIdx.x) * 4;
  float4 v = *(const float4*)(src + i);
  union { unsigned short s[4]; uint2 u; } o;
  o.s[0] = f2bf(v.x); o.s[1] = f2bf(v.y); o.s[2] = f2bf(v.z); o.s[3] = f2bf(v.w);
  *(uint2*)(dst + i) = o.u;
}

// ---------- Kernel 1: weight transpose + convert: Wt[n][d] = W[d][n] ----------
__global__ __launch_bounds__(256) void k_transW(const float* __restrict__ W,
                                                unsigned short* __restrict__ Wt) {
  int idx = blockIdx.x * 256 + threadIdx.x;
  int d = idx >> 10, n = idx & (DD - 1);
  Wt[(size_t)n * DD + d] = f2bf(W[idx]);
}

// ---------- Kernel 2: GEMM + bias, 128x128 block tile, async double-buffer ----------
// A [4096][1024] bf16 row-major; Bt [N=1024][K=1024] bf16 (pre-transposed W)
__device__ __forceinline__ void gemm_stage(unsigned short* as, unsigned short* bs,
                                           const unsigned short* A,
                                           const unsigned short* Bt,
                                           int m0, int n0, int k0, int tid) {
#pragma unroll
  for (int i = 0; i < 2; ++i) {
    int c = tid + 256 * i;
    int row = c >> 2, seg = c & 3;
    async_b128(as + row * 40 + seg * 8, A + (size_t)(m0 + row) * DD + k0 + seg * 8);
    async_b128(bs + row * 40 + seg * 8, Bt + (size_t)(n0 + row) * DD + k0 + seg * 8);
  }
}

__global__ __launch_bounds__(256) void k_gemm(const unsigned short* __restrict__ A,
                                              const unsigned short* __restrict__ Bt,
                                              const float* __restrict__ bias,
                                              unsigned short* __restrict__ C) {
  const int m0 = blockIdx.x * 128, n0 = blockIdx.y * 128;
  const int tid = threadIdx.x;
  const int w = tid >> 5, lane = tid & 31, ln = lane & 15, hf = lane >> 4;
  __shared__ unsigned short As[2][128 * 40];  // [m][k], stride 40 bf16
  __shared__ unsigned short Bs[2][128 * 40];  // [n][k]
  v8f acc[8];
#pragma unroll
  for (int t = 0; t < 8; ++t) acc[t] = zero_v8f();

  gemm_stage(As[0], Bs[0], A, Bt, m0, n0, 0, tid);
  wait_async0();
  __syncthreads();

  for (int it = 0; it < DD / 32; ++it) {
    int cur = it & 1;
    if (it + 1 < DD / 32)
      gemm_stage(As[cur ^ 1], Bs[cur ^ 1], A, Bt, m0, n0, (it + 1) * 32, tid);
    v16bf af = load_a_frag(As[cur], 40, 16 * w + ln, 0);
#pragma unroll
    for (int t = 0; t < 8; ++t) {
      v16bf bfr = load_b_frag(Bs[cur], 40, 16 * t + ln, 0);
      acc[t] = wmma_bf16(af, bfr, acc[t]);
    }
    wait_async0();
    __syncthreads();
  }
#pragma unroll
  for (int t = 0; t < 8; ++t) {
    int n = n0 + 16 * t + ln;
    float bv = bias[n];
#pragma unroll
    for (int r = 0; r < 8; ++r) {
      int m = m0 + 16 * w + r + 8 * hf;
      C[(size_t)m * DD + n] = f2bf(acc[t][r] + bv);
    }
  }
}

// ---------- Kernel 3: column-softmax stats over q: M[k], Z[k] ----------
// T[k,q] = (key . ctx)/8 + mask[b,q,k]; softmax axis is q => row reduce of T.
__global__ __launch_bounds__(256) void k_stats(const unsigned short* __restrict__ ctxb,
                                               const unsigned short* __restrict__ keyb,
                                               const float* __restrict__ mask,
                                               float* __restrict__ gM,
                                               float* __restrict__ gZ) {
  const int bb = blockIdx.z, hh = blockIdx.y;
  const int k0 = blockIdx.x * 128;
  const int tid = threadIdx.x;
  const int w = tid >> 5, lane = tid & 31, ln = lane & 15, hf = lane >> 4;
  __shared__ unsigned short Ks[128 * 72];     // key tile [k][d], resident
  __shared__ unsigned short Cs[2][128 * 72];  // ctx tile [q][d], double-buffered
#pragma unroll
  for (int i = 0; i < 4; ++i) {
    int c = tid + 256 * i;
    int row = c >> 3, seg = c & 7;
    async_b128(&Ks[row * 72 + seg * 8],
               &keyb[((size_t)(bb * SS + k0 + row)) * DD + hh * HDIM + seg * 8]);
    async_b128(&Cs[0][row * 72 + seg * 8],
               &ctxb[((size_t)(bb * SS + row)) * DD + hh * HDIM + seg * 8]);
  }
  wait_async0();
  __syncthreads();
  v16bf a0 = load_a_frag(Ks, 72, 16 * w + ln, 0);   // cached in VGPRs
  v16bf a1 = load_a_frag(Ks, 72, 16 * w + ln, 32);
  float Mreg[8], st[8];
  // 32 flat iterations = 2 sweeps (max, sum-exp) x 16 q-tiles, prefetch flows across
  for (int it = 0; it < 32; ++it) {
    const int sweep = it >> 4, qt = it & 15, cur = it & 1;
    if (qt == 0) {
#pragma unroll
      for (int r = 0; r < 8; ++r) st[r] = sweep ? 0.0f : -3.0e38f;
    }
    if (it + 1 < 32) {
      const int qn = (it + 1) & 15;
#pragma unroll
      for (int i = 0; i < 4; ++i) {
        int c = tid + 256 * i;
        int row = c >> 3, seg = c & 7;
        async_b128(&Cs[cur ^ 1][row * 72 + seg * 8],
                   &ctxb[((size_t)(bb * SS + qn * 128 + row)) * DD + hh * HDIM +
                         seg * 8]);
      }
    }
    for (int qf = 0; qf < 8; ++qf) {
      v16bf b0 = load_b_frag(Cs[cur], 72, 16 * qf + ln, 0);
      v16bf b1 = load_b_frag(Cs[cur], 72, 16 * qf + ln, 32);
      v8f c8 = zero_v8f();
      c8 = wmma_bf16(a0, b0, c8);
      c8 = wmma_bf16(a1, b1, c8);
      int q = qt * 128 + 16 * qf + ln;
      const float* mp = mask + ((size_t)bb * SS + q) * SS + k0 + 16 * w + 8 * hf;
      float4 mA = *(const float4*)mp;
      float4 mB = *(const float4*)(mp + 4);
      float mv[8] = {mA.x, mA.y, mA.z, mA.w, mB.x, mB.y, mB.z, mB.w};
      if (sweep == 0) {
#pragma unroll
        for (int r = 0; r < 8; ++r)
          st[r] = fmaxf(st[r], c8[r] * 0.125f + mv[r]);
      } else {
#pragma unroll
        for (int r = 0; r < 8; ++r)
          st[r] += __expf(c8[r] * 0.125f + mv[r] - Mreg[r]);
      }
    }
    wait_async0();
    __syncthreads();
    if (qt == 15) {  // reduce over the 16 q-lanes of each half-wave
#pragma unroll
      for (int off = 1; off < 16; off <<= 1) {
#pragma unroll
        for (int r = 0; r < 8; ++r) {
          float o = __shfl_xor(st[r], off, 32);
          st[r] = sweep ? (st[r] + o) : fmaxf(st[r], o);
        }
      }
      if (sweep == 0) {
#pragma unroll
        for (int r = 0; r < 8; ++r) Mreg[r] = st[r];
      }
    }
  }
  if (ln == 0) {
    size_t base = ((size_t)bb * HH + hh) * SS + k0 + 16 * w + 8 * hf;
#pragma unroll
    for (int r = 0; r < 8; ++r) { gM[base + r] = Mreg[r]; gZ[base + r] = st[r]; }
  }
}

// ---------- Kernel 4: valT[b,h,d,k] = val[b,k,h*64+d] / Z[b,h,k] ----------
__global__ __launch_bounds__(256) void k_scaleT(const unsigned short* __restrict__ valb,
                                                const float* __restrict__ gZ,
                                                unsigned short* __restrict__ valT) {
  const int bb = blockIdx.z, hh = blockIdx.y;
  const int k0 = blockIdx.x * 64;
  const int tid = threadIdx.x;
  __shared__ unsigned short T[64 * 72];  // [k][d]
#pragma unroll
  for (int i = 0; i < 2; ++i) {
    int c = tid + 256 * i;
    int row = c >> 3, seg = c & 7;
    *(uint4*)&T[row * 72 + seg * 8] =
        *(const uint4*)&valb[((size_t)(bb * SS + k0 + row)) * DD + hh * HDIM + seg * 8];
  }
  __syncthreads();
#pragma unroll
  for (int i = 0; i < 2; ++i) {
    int c = tid + 256 * i;
    int d = c >> 3, seg = c & 7;
    size_t zb = ((size_t)bb * HH + hh) * SS + k0 + seg * 8;
    float4 zA = *(const float4*)(gZ + zb);
    float4 zB = *(const float4*)(gZ + zb + 4);
    float zz[8] = {zA.x, zA.y, zA.z, zA.w, zB.x, zB.y, zB.z, zB.w};
    union { unsigned short s[8]; uint4 u; } o;
#pragma unroll
    for (int j = 0; j < 8; ++j)
      o.s[j] = f2bf(bf2f(T[(seg * 8 + j) * 72 + d]) / zz[j]);
    *(uint4*)&valT[(((size_t)bb * HH + hh) * HDIM + d) * SS + k0 + seg * 8] = o.u;
  }
}

// ---------- Kernel 5: out^T[d,q] = sum_k valT[d,k] * exp(T[k,q]-M[k]) ----------
__global__ __launch_bounds__(256) void k_attn_out(const unsigned short* __restrict__ ctxb,
                                                  const unsigned short* __restrict__ keyb,
                                                  const unsigned short* __restrict__ valT,
                                                  const float* __restrict__ mask,
                                                  const float* __restrict__ gM,
                                                  float* __restrict__ out) {
  const int bb = blockIdx.z, hh = blockIdx.y;
  const int q0 = blockIdx.x * 128;
  const int tid = threadIdx.x;
  const int w = tid >> 5, lane = tid & 31, ln = lane & 15, hf = lane >> 4;
  __shared__ unsigned short Cs[128 * 72];     // ctx tile [q][d], resident
  __shared__ unsigned short Kt[2][32 * 72];   // key tile [k][d], double-buffered
  __shared__ unsigned short Vt[2][64 * 40];   // valT tile [d][k], double-buffered
  __shared__ unsigned short Ps[128 * 40];     // p tile [q][k] (per-wave rows only)
  const size_t keybase = (size_t)bb * SS * DD + hh * HDIM;
  const size_t vbase = ((size_t)bb * HH + hh) * HDIM * SS;
#pragma unroll
  for (int i = 0; i < 4; ++i) {
    int c = tid + 256 * i;
    int row = c >> 3, seg = c & 7;
    async_b128(&Cs[row * 72 + seg * 8],
               &ctxb[((size_t)(bb * SS + q0 + row)) * DD + hh * HDIM + seg * 8]);
  }
  async_b128(&Kt[0][(tid >> 3) * 72 + (tid & 7) * 8],
             &keyb[keybase + (size_t)(tid >> 3) * DD + (tid & 7) * 8]);
  async_b128(&Vt[0][(tid >> 2) * 40 + (tid & 3) * 8],
             &valT[vbase + (size_t)(tid >> 2) * SS + (tid & 3) * 8]);
  wait_async0();
  __syncthreads();
  v16bf bq0 = load_b_frag(Cs, 72, 16 * w + ln, 0);   // cached for whole k loop
  v16bf bq1 = load_b_frag(Cs, 72, 16 * w + ln, 32);
  v8f acc[4];
#pragma unroll
  for (int t = 0; t < 4; ++t) acc[t] = zero_v8f();
  const int q = q0 + 16 * w + ln;

  for (int kt = 0; kt < SS / 32; ++kt) {
    const int cur = kt & 1;
    if (kt + 1 < SS / 32) {
      async_b128(&Kt[cur ^ 1][(tid >> 3) * 72 + (tid & 7) * 8],
                 &keyb[keybase + (size_t)((kt + 1) * 32 + (tid >> 3)) * DD +
                       (tid & 7) * 8]);
      async_b128(&Vt[cur ^ 1][(tid >> 2) * 40 + (tid & 3) * 8],
                 &valT[vbase + (size_t)(tid >> 2) * SS + (kt + 1) * 32 +
                       (tid & 3) * 8]);
    }
#pragma unroll
    for (int kk = 0; kk < 2; ++kk) {
      v16bf ak0 = load_a_frag(Kt[cur], 72, 16 * kk + ln, 0);
      v16bf ak1 = load_a_frag(Kt[cur], 72, 16 * kk + ln, 32);
      v8f c8 = zero_v8f();
      c8 = wmma_bf16(ak0, bq0, c8);
      c8 = wmma_bf16(ak1, bq1, c8);
      int kb = kt * 32 + 16 * kk + 8 * hf;
      const float* mp = mask + ((size_t)bb * SS + q) * SS + kb;
      const float* Mp = gM + ((size_t)bb * HH + hh) * SS + kb;
      float4 mA = *(const float4*)mp, mB = *(const float4*)(mp + 4);
      float4 MA = *(const float4*)Mp, MB = *(const float4*)(Mp + 4);
      float mv[8] = {mA.x, mA.y, mA.z, mA.w, mB.x, mB.y, mB.z, mB.w};
      float Mv[8] = {MA.x, MA.y, MA.z, MA.w, MB.x, MB.y, MB.z, MB.w};
      union { unsigned short s[8]; uint4 u; } pk;
#pragma unroll
      for (int r = 0; r < 8; ++r)
        pk.s[r] = f2bf(__expf(c8[r] * 0.125f + mv[r] - Mv[r]));
      // per-wave rows of Ps; DS ops are in-order within a wave, so the
      // B-fragment reads below see this data without a workgroup barrier
      *(uint4*)&Ps[(16 * w + ln) * 40 + 16 * kk + 8 * hf] = pk.u;
    }
    v16bf bp = load_b_frag(Ps, 40, 16 * w + ln, 0);
#pragma unroll
    for (int t = 0; t < 4; ++t) {
      v16bf av = load_a_frag(Vt[cur], 40, 16 * t + ln, 0);
      acc[t] = wmma_bf16(av, bp, acc[t]);
    }
    wait_async0();
    __syncthreads();
  }
  // Epilogue: C rows are d (contiguous in memory) -> two float4 stores per frag
#pragma unroll
  for (int t = 0; t < 4; ++t) {
    float4 lo, hi;
    lo.x = acc[t][0]; lo.y = acc[t][1]; lo.z = acc[t][2]; lo.w = acc[t][3];
    hi.x = acc[t][4]; hi.y = acc[t][5]; hi.z = acc[t][6]; hi.w = acc[t][7];
    float* op = out + ((size_t)bb * SS + q) * DD + hh * HDIM + 16 * t + 8 * hf;
    *(float4*)op = lo;
    *(float4*)(op + 4) = hi;
  }
}

// ---------- Host launcher ----------
extern "C" void kernel_launch(void* const* d_in, const int* in_sizes, int n_in,
                              void* d_out, int out_size, void* d_ws, size_t ws_size,
                              hipStream_t stream) {
  const float* features = (const float*)d_in[0];
  const float* hiddens  = (const float*)d_in[1];
  const float* mask     = (const float*)d_in[2];
  const float* Wc = (const float*)d_in[3];
  const float* bc = (const float*)d_in[4];
  const float* Wk = (const float*)d_in[5];
  const float* bk = (const float*)d_in[6];
  const float* Wv = (const float*)d_in[7];
  const float* bv = (const float*)d_in[8];
  float* out = (float*)d_out;

  char* ws = (char*)d_ws;
  const size_t SZ_BSD_BF = (size_t)BB * SS * DD * 2;  // 8 MB
  const size_t SZ_W_BF   = (size_t)DD * DD * 2;       // 2 MB
  const size_t SZ_STAT   = (size_t)BB * HH * SS * 4;  // 256 KB
  size_t o = 0;
  unsigned short* fb   = (unsigned short*)(ws + o); o += SZ_BSD_BF;
  unsigned short* hb   = (unsigned short*)(ws + o); o += SZ_BSD_BF;
  unsigned short* WcT  = (unsigned short*)(ws + o); o += SZ_W_BF;
  unsigned short* WkT  = (unsigned short*)(ws + o); o += SZ_W_BF;
  unsigned short* WvT  = (unsigned short*)(ws + o); o += SZ_W_BF;
  unsigned short* ctxb = (unsigned short*)(ws + o); o += SZ_BSD_BF;
  unsigned short* keyb = (unsigned short*)(ws + o); o += SZ_BSD_BF;
  unsigned short* valb = (unsigned short*)(ws + o); o += SZ_BSD_BF;
  float*          gM   = (float*)(ws + o);          o += SZ_STAT;
  float*          gZ   = (float*)(ws + o);          o += SZ_STAT;
  unsigned short* valT = (unsigned short*)(ws + o); o += SZ_BSD_BF;

  const int nBSD = BB * SS * DD;  // 4194304
  k_cvt<<<nBSD / 1024, 256, 0, stream>>>(features, fb);
  k_cvt<<<nBSD / 1024, 256, 0, stream>>>(hiddens, hb);
  k_transW<<<(DD * DD) / 256, 256, 0, stream>>>(Wc, WcT);
  k_transW<<<(DD * DD) / 256, 256, 0, stream>>>(Wk, WkT);
  k_transW<<<(DD * DD) / 256, 256, 0, stream>>>(Wv, WvT);

  dim3 gg(BB * SS / 128, DD / 128);  // 32 x 8
  k_gemm<<<gg, 256, 0, stream>>>(fb, WcT, bc, ctxb);
  k_gemm<<<gg, 256, 0, stream>>>(fb, WkT, bk, keyb);
  k_gemm<<<gg, 256, 0, stream>>>(hb, WvT, bv, valb);

  k_stats<<<dim3(SS / 128, HH, BB), 256, 0, stream>>>(ctxb, keyb, mask, gM, gZ);
  k_scaleT<<<dim3(SS / 64, HH, BB), 256, 0, stream>>>(valb, gZ, valT);
  k_attn_out<<<dim3(SS / 128, HH, BB), 256, 0, stream>>>(ctxb, keyb, valT, mask, gM, out);
}
